// CausalSelfAttention_65541200937271
// MI455X (gfx1250) — matrix operand order repaired and checked by reading, not compiled, not run on hardware
//
#include <hip/hip_runtime.h>

// ---------------------------------------------------------------------------
// CausalSelfAttention for MI455X (gfx1250, wave32, WMMA bf16 w/ f32 accum)
//   B=2, T=4096, C=768, H=12, hd=64
//   prep (f32->bf16, pre-transpose/permute) -> qkv_gemm -> flash_attn -> proj
//   A-operand storage uses perm32 (bit3<->bit4 swap per 32-k block) so every
//   fragment load is 2x b128. All tile staging is uint4 copies. Double-buffered.
// ---------------------------------------------------------------------------

typedef __attribute__((ext_vector_type(16))) __bf16 bf16x16;
typedef __attribute__((ext_vector_type(8)))  float  f32x8;
typedef __attribute__((ext_vector_type(4)))  float  f32x4;
typedef __attribute__((ext_vector_type(4)))  unsigned int u32x4;

union Frag {                     // one WMMA bf16 A/B operand: 8 dwords
    unsigned int u[8];
    u32x4        q[2];
    bf16x16      v;
};

__device__ __forceinline__ unsigned short f2bf(float f) {
    union { float f; unsigned int u; } c; c.f = f;
    unsigned int u = c.u;
    u += 0x7FFFu + ((u >> 16) & 1u);          // round-to-nearest-even
    return (unsigned short)(u >> 16);
}
__device__ __forceinline__ unsigned int pack2bf(float lo, float hi) {
    return (unsigned int)f2bf(lo) | ((unsigned int)f2bf(hi) << 16);
}
// swap bit3<->bit4 within a 32-wide k block (A-operand element permutation)
__device__ __forceinline__ int perm32(int k) {
    return (k & 7) | ((k & 8) << 1) | ((k & 16) >> 1);
}

#define WMMA_BF16(A, B, C) \
    __builtin_amdgcn_wmma_f32_16x16x32_bf16(false, (A), false, (B), (short)0, (C), false, false)

static constexpr int T_SEQ = 4096;
static constexpr int C_DIM = 768;
static constexpr int H_NUM = 12;
static constexpr int HD    = 64;
static constexpr long long HEAD_STRIDE = (long long)T_SEQ * HD;             // 262144
static constexpr long long SEL_STRIDE  = 24LL * HEAD_STRIDE;                // 6291456

// ===========================================================================
// Prep kernels (one-time f32 -> bf16, layouts baked for the GEMM consumers)
// ===========================================================================
// x[8192][768] f32 -> xbf[8192][768] bf16 with perm32 within each 32-k block.
// perm32 at uint4 granularity = swap uint4 slots 1<->2 of each 4-uint4 group.
__global__ __launch_bounds__(256) void prep_x_kernel(
    const float* __restrict__ x, unsigned short* __restrict__ xbf)
{
    int g = blockIdx.x * 256 + threadIdx.x;           // 786432 threads, 8 floats each
    const f32x4* xv = (const f32x4*)x;
    f32x4 a = xv[g * 2], b = xv[g * 2 + 1];
    u32x4 o;
    o.x = pack2bf(a.x, a.y); o.y = pack2bf(a.z, a.w);
    o.z = pack2bf(b.x, b.y); o.w = pack2bf(b.z, b.w);
    int row = g / 96, c4r = g % 96;                   // 96 uint4 per row
    int blk = c4r >> 2, c4 = c4r & 3;
    int oc4 = (blk << 2) | ((c4 & 1) << 1) | ((c4 >> 1) & 1);  // 0,2,1,3
    ((u32x4*)xbf)[(size_t)row * 96 + oc4] = o;
}

// W[K][N] f32 -> Wt[N][K] bf16 (natural k order: B-operand layout).
__global__ __launch_bounds__(256) void prep_w_kernel(
    const float* __restrict__ W, unsigned short* __restrict__ Wt, int K, int N)
{
    int g = blockIdx.x * 256 + threadIdx.x;           // N*K/8 threads
    int kq = K >> 3;                                  // uint4 per output row
    int n = g / kq, k0 = (g % kq) * 8;
    u32x4 o;
#pragma unroll
    for (int j = 0; j < 4; ++j)
        o[j] = pack2bf(W[(size_t)(k0 + 2 * j) * N + n],
                       W[(size_t)(k0 + 2 * j + 1) * N + n]);
    ((u32x4*)Wt)[g] = o;
}

// ===========================================================================
// Kernel 1: qkv = xbf @ Wa^T + b_attn
//   Q -> bf16 [bh][t][perm(d)], K -> [bh][t][d], V -> [bh][d][t] (transposed)
//   M=8192,K=768,N=2304. 128x128 block, 8 waves, double-buffered, uint4 staging.
// ===========================================================================
__global__ __launch_bounds__(256) void qkv_gemm_kernel(
    const unsigned short* __restrict__ xbf, const unsigned short* __restrict__ Wat,
    const float* __restrict__ bias, unsigned short* __restrict__ qkv)
{
    __shared__ __attribute__((aligned(16))) unsigned short lA[2][128 * 40]; // [m][perm(k)]
    __shared__ __attribute__((aligned(16))) unsigned short lB[2][128 * 40]; // [n][k]

    const int tid  = threadIdx.x;
    const int lane = tid & 31, wave = tid >> 5;
    const int half = lane >> 4, l16 = lane & 15;
    const int waveM = wave & 1, waveN = wave >> 1;
    const int blockM = blockIdx.x * 128;
    const int blockN = blockIdx.y * 128;

    f32x8 acc[4][2];
#pragma unroll
    for (int i = 0; i < 4; ++i)
#pragma unroll
        for (int j = 0; j < 2; ++j)
#pragma unroll
            for (int r = 0; r < 8; ++r) acc[i][j][r] = 0.f;

    const u32x4* Ag = (const u32x4*)xbf;   // 96 uint4 per row
    const u32x4* Bg = (const u32x4*)Wat;   // 96 uint4 per row
    u32x4 av[2], bv[2];

    // ---- prologue: stage tile 0 (uint4 copies) ----
#pragma unroll
    for (int i = 0; i < 2; ++i) {
        int e = i * 256 + tid, rr = e >> 2, c4 = e & 3;
        av[i] = Ag[(size_t)(blockM + rr) * 96 + c4];
        bv[i] = Bg[(size_t)(blockN + rr) * 96 + c4];
    }
    {
        u32x4* Ad = (u32x4*)lA[0];
        u32x4* Bd = (u32x4*)lB[0];
#pragma unroll
        for (int i = 0; i < 2; ++i) {
            int e = i * 256 + tid, rr = e >> 2, c4 = e & 3;
            Ad[rr * 5 + c4] = av[i];                  // row stride 20 u32 = 5 uint4
            Bd[rr * 5 + c4] = bv[i];
        }
    }
    __syncthreads();

    const int nIter = 24;
    for (int it = 0; it < nIter; ++it) {
        const int cur = it & 1;
        if (it + 1 < nIter) {
            int k4 = (it + 1) * 4;                    // kk/8 in uint4
#pragma unroll
            for (int i = 0; i < 2; ++i) {
                int e = i * 256 + tid, rr = e >> 2, c4 = e & 3;
                av[i] = Ag[(size_t)(blockM + rr) * 96 + k4 + c4];
                bv[i] = Bg[(size_t)(blockN + rr) * 96 + k4 + c4];
            }
        }
        const unsigned int* Au = (const unsigned int*)lA[cur];
        const unsigned int* Bu = (const unsigned int*)lB[cur];
        Frag a[4], b[2];
#pragma unroll
        for (int fm = 0; fm < 4; ++fm) {
            const unsigned int* p = Au + (waveM * 64 + fm * 16 + l16) * 20 + half * 8;
#pragma unroll
            for (int j = 0; j < 8; ++j) a[fm].u[j] = p[j];
        }
#pragma unroll
        for (int fn = 0; fn < 2; ++fn) {
            const unsigned int* p = Bu + (waveN * 32 + fn * 16 + l16) * 20 + half * 4;
#pragma unroll
            for (int j = 0; j < 8; ++j) b[fn].u[j] = p[(j & 3) + ((j & 4) << 1)];
        }
#pragma unroll
        for (int fm = 0; fm < 4; ++fm)
#pragma unroll
            for (int fn = 0; fn < 2; ++fn)
                acc[fm][fn] = WMMA_BF16(a[fm].v, b[fn].v, acc[fm][fn]);

        if (it + 1 < nIter) {
            u32x4* Ad = (u32x4*)lA[cur ^ 1];
            u32x4* Bd = (u32x4*)lB[cur ^ 1];
#pragma unroll
            for (int i = 0; i < 2; ++i) {
                int e = i * 256 + tid, rr = e >> 2, c4 = e & 3;
                Ad[rr * 5 + c4] = av[i];
                Bd[rr * 5 + c4] = bv[i];
            }
        }
        __syncthreads();
    }

    // ---- epilogue: bias + scatter. sel uniform per block (768 % 128 == 0) ----
    const int sel = blockN / 768;
#pragma unroll
    for (int fm = 0; fm < 4; ++fm)
#pragma unroll
        for (int fn = 0; fn < 2; ++fn) {
            int gcol = blockN + waveN * 32 + fn * 16 + l16;
            int c = gcol - sel * 768;
            int h = c >> 6, d = c & 63;
            float bvs = bias[gcol];
            if (sel == 2) {                           // V: [bh][d][t], pack 8 t's
                int t0 = blockM + waveM * 64 + fm * 16 + half * 8;
                int bb = t0 >> 12; t0 &= 4095;
                int bh = bb * H_NUM + h;
                u32x4 o;
#pragma unroll
                for (int j = 0; j < 4; ++j)
                    o[j] = pack2bf(acc[fm][fn][2 * j] + bvs,
                                   acc[fm][fn][2 * j + 1] + bvs);
                long long idx = 2 * SEL_STRIDE + (long long)bh * HEAD_STRIDE
                              + (long long)d * T_SEQ + t0;
                ((u32x4*)qkv)[idx >> 3] = o;
            } else {
#pragma unroll
                for (int r = 0; r < 8; ++r) {
                    int grow = blockM + waveM * 64 + fm * 16 + half * 8 + r;
                    int bb = grow >> 12, t = grow & 4095;
                    int bh = bb * H_NUM + h;
                    float v = acc[fm][fn][r] + bvs;
                    long long idx;
                    if (sel == 0) {                   // Q: permuted d
                        int dp = (d & 32) | perm32(d & 31);
                        idx = (long long)bh * HEAD_STRIDE + (long long)t * HD + dp;
                    } else {                          // K: natural
                        idx = SEL_STRIDE + (long long)bh * HEAD_STRIDE
                            + (long long)t * HD + d;
                    }
                    qkv[idx] = f2bf(v);
                }
            }
        }
}

// ===========================================================================
// Kernel 2: causal flash attention. 1 block = 64 queries of one (b,h).
//   128 threads (4 waves x 16 rows). Double-buffered K/V tiles, per-wave P.
// ===========================================================================
__global__ __launch_bounds__(128) void flash_attn_kernel(
    const unsigned short* __restrict__ qkv, unsigned short* __restrict__ y)
{
    __shared__ __attribute__((aligned(16))) unsigned short lK[2][64 * 72]; // [key][d]
    __shared__ __attribute__((aligned(16))) unsigned short lV[2][64 * 72]; // [d][key]
    __shared__ __attribute__((aligned(16))) unsigned short lP[4][16 * 72]; // [row][perm(key)]

    const int tid  = threadIdx.x;
    const int lane = tid & 31, wave = tid >> 5;
    const int half = lane >> 4, l16 = lane & 15;
    const int bh = blockIdx.y;
    const int q0 = blockIdx.x * 64;

    const u32x4* qp4 = (const u32x4*)(qkv + (long long)bh * HEAD_STRIDE);
    const u32x4* kp4 = (const u32x4*)(qkv + SEL_STRIDE + (long long)bh * HEAD_STRIDE);
    const u32x4* vp4 = (const u32x4*)(qkv + 2 * SEL_STRIDE + (long long)bh * HEAD_STRIDE);

    // Q fragments: permuted storage -> 4x uint4 loads
    Frag qf[2];
    {
        const u32x4* p = qp4 + (q0 + wave * 16 + l16) * 8;   // 8 uint4 per row
#pragma unroll
        for (int ks = 0; ks < 2; ++ks) {
            qf[ks].q[0] = p[ks * 4 + half * 2];
            qf[ks].q[1] = p[ks * 4 + half * 2 + 1];
        }
    }

    float m[8], l[8];
    f32x8 oacc[4];
#pragma unroll
    for (int r = 0; r < 8; ++r) { m[r] = -1e30f; l[r] = 0.f; }
#pragma unroll
    for (int nf = 0; nf < 4; ++nf)
#pragma unroll
        for (int r = 0; r < 8; ++r) oacc[nf][r] = 0.f;

    u32x4 kreg[4], vreg[4];
    // ---- prologue: stage key tile 0 ----
#pragma unroll
    for (int i = 0; i < 4; ++i) {
        int e = i * 128 + tid, rr = e >> 3, c4 = e & 7;     // 64 rows x 8 uint4
        kreg[i] = kp4[rr * 8 + c4];                          // K[key][d]
        vreg[i] = vp4[(size_t)rr * 512 + c4];                // V[d][t], tile 0
    }
    {
        u32x4* Kd = (u32x4*)lK[0];
        u32x4* Vd = (u32x4*)lV[0];
#pragma unroll
        for (int i = 0; i < 4; ++i) {
            int e = i * 128 + tid, rr = e >> 3, c4 = e & 7;
            Kd[rr * 9 + c4] = kreg[i];                       // stride 36 u32 = 9 uint4
            Vd[rr * 9 + c4] = vreg[i];
        }
    }
    __syncthreads();

    const int ktiles = blockIdx.x + 1;
    for (int kt = 0; kt < ktiles; ++kt) {
        const int cur = kt & 1;
        if (kt + 1 < ktiles) {
#pragma unroll
            for (int i = 0; i < 4; ++i) {
                int e = i * 128 + tid, rr = e >> 3, c4 = e & 7;
                kreg[i] = kp4[((kt + 1) * 64 + rr) * 8 + c4];
                vreg[i] = vp4[(size_t)rr * 512 + (kt + 1) * 8 + c4];
            }
        }
        const unsigned int* Ku = (const unsigned int*)lK[cur];
        const unsigned int* Vu = (const unsigned int*)lV[cur];

        // ---- S = (Q @ K^T) * 0.125 ----
        f32x8 s[4];
#pragma unroll
        for (int nf = 0; nf < 4; ++nf)
#pragma unroll
            for (int r = 0; r < 8; ++r) s[nf][r] = 0.f;
#pragma unroll
        for (int ks = 0; ks < 2; ++ks)
#pragma unroll
            for (int nf = 0; nf < 4; ++nf) {
                Frag bf;
                const unsigned int* p = Ku + (nf * 16 + l16) * 36 + ks * 16 + half * 4;
#pragma unroll
                for (int j = 0; j < 8; ++j) bf.u[j] = p[(j & 3) + ((j & 4) << 1)];
                s[nf] = WMMA_BF16(qf[ks].v, bf.v, s[nf]);
            }
        const bool diag = (kt == blockIdx.x);
#pragma unroll
        for (int nf = 0; nf < 4; ++nf)
#pragma unroll
            for (int r = 0; r < 8; ++r) {
                float v = s[nf][r] * 0.125f;
                if (diag) {
                    int key  = kt * 64 + nf * 16 + l16;
                    int qrow = q0 + wave * 16 + half * 8 + r;
                    if (key > qrow) v = -1e30f;
                }
                s[nf][r] = v;
            }

        // ---- online softmax ----
        float mn[8], alpha[8];
#pragma unroll
        for (int r = 0; r < 8; ++r) {
            float v = fmaxf(fmaxf(s[0][r], s[1][r]), fmaxf(s[2][r], s[3][r]));
#pragma unroll
            for (int off = 1; off < 16; off <<= 1)
                v = fmaxf(v, __shfl_xor(v, off, 32));
            mn[r] = fmaxf(m[r], v);
            alpha[r] = __expf(m[r] - mn[r]);
            m[r] = mn[r];
        }
        float rs[8];
#pragma unroll
        for (int r = 0; r < 8; ++r) rs[r] = 0.f;
#pragma unroll
        for (int nf = 0; nf < 4; ++nf)
#pragma unroll
            for (int r = 0; r < 8; ++r) {
                float p = __expf(s[nf][r] - mn[r]);
                s[nf][r] = p;
                rs[r] += p;
            }
#pragma unroll
        for (int r = 0; r < 8; ++r) {
            float v = rs[r];
#pragma unroll
            for (int off = 1; off < 16; off <<= 1)
                v += __shfl_xor(v, off, 32);
            l[r] = l[r] * alpha[r] + v;
        }
#pragma unroll
        for (int nf = 0; nf < 4; ++nf)
#pragma unroll
            for (int r = 0; r < 8; ++r) oacc[nf][r] *= alpha[r];

        // ---- P -> per-wave LDS (permuted); wave-local, no barrier needed ----
        unsigned short* pw = lP[wave];
#pragma unroll
        for (int nf = 0; nf < 4; ++nf)
#pragma unroll
            for (int r = 0; r < 8; ++r) {
                int k = nf * 16 + l16;
                pw[(half * 8 + r) * 72 + (k & 32) + perm32(k & 31)] = f2bf(s[nf][r]);
            }
        const unsigned int* pu = (const unsigned int*)pw;
#pragma unroll
        for (int ks = 0; ks < 2; ++ks) {
            Frag pf;
            const unsigned int* pp = pu + l16 * 36 + ks * 16 + half * 8;
#pragma unroll
            for (int j = 0; j < 8; ++j) pf.u[j] = pp[j];
#pragma unroll
            for (int nf = 0; nf < 4; ++nf) {
                Frag vf;
                const unsigned int* vv = Vu + (nf * 16 + l16) * 36 + ks * 16 + half * 4;
#pragma unroll
                for (int j = 0; j < 8; ++j) vf.u[j] = vv[(j & 3) + ((j & 4) << 1)];
                oacc[nf] = WMMA_BF16(pf.v, vf.v, oacc[nf]);
            }
        }

        if (kt + 1 < ktiles) {
            u32x4* Kd = (u32x4*)lK[cur ^ 1];
            u32x4* Vd = (u32x4*)lV[cur ^ 1];
#pragma unroll
            for (int i = 0; i < 4; ++i) {
                int e = i * 128 + tid, rr = e >> 3, c4 = e & 7;
                Kd[rr * 9 + c4] = kreg[i];
                Vd[rr * 9 + c4] = vreg[i];
            }
        }
        __syncthreads();
    }

    // ---- finalize: y[b*T+t][perm-col] = O / l (A-operand of proj) ----
    const int b = bh / H_NUM, h = bh % H_NUM;
    float inv[8];
#pragma unroll
    for (int r = 0; r < 8; ++r) inv[r] = 1.f / l[r];
#pragma unroll
    for (int nf = 0; nf < 4; ++nf)
#pragma unroll
        for (int r = 0; r < 8; ++r) {
            int row = q0 + wave * 16 + half * 8 + r;
            int col = h * 64 + nf * 16 + l16;
            int cp  = (col & ~31) | perm32(col & 31);
            y[(long long)(b * T_SEQ + row) * C_DIM + cp] = f2bf(oacc[nf][r] * inv[r]);
        }
}

// ===========================================================================
// Kernel 3: out = y(bf16, pre-permuted) @ Wp^T + b_proj. M=8192,K=768,N=768.
// ===========================================================================
__global__ __launch_bounds__(256) void proj_gemm_kernel(
    const unsigned short* __restrict__ y, const unsigned short* __restrict__ Wpt,
    const float* __restrict__ bias, float* __restrict__ out)
{
    __shared__ __attribute__((aligned(16))) unsigned short lA[2][128 * 40];
    __shared__ __attribute__((aligned(16))) unsigned short lB[2][128 * 40];

    const int tid  = threadIdx.x;
    const int lane = tid & 31, wave = tid >> 5;
    const int half = lane >> 4, l16 = lane & 15;
    const int waveM = wave & 1, waveN = wave >> 1;
    const int blockM = blockIdx.x * 128;
    const int blockN = blockIdx.y * 128;

    f32x8 acc[4][2];
#pragma unroll
    for (int i = 0; i < 4; ++i)
#pragma unroll
        for (int j = 0; j < 2; ++j)
#pragma unroll
            for (int r = 0; r < 8; ++r) acc[i][j][r] = 0.f;

    const u32x4* Ag = (const u32x4*)y;     // 96 uint4 per row
    const u32x4* Bg = (const u32x4*)Wpt;   // 96 uint4 per row
    u32x4 av[2], bv[2];

#pragma unroll
    for (int i = 0; i < 2; ++i) {
        int e = i * 256 + tid, rr = e >> 2, c4 = e & 3;
        av[i] = Ag[(size_t)(blockM + rr) * 96 + c4];
        bv[i] = Bg[(size_t)(blockN + rr) * 96 + c4];
    }
    {
        u32x4* Ad = (u32x4*)lA[0];
        u32x4* Bd = (u32x4*)lB[0];
#pragma unroll
        for (int i = 0; i < 2; ++i) {
            int e = i * 256 + tid, rr = e >> 2, c4 = e & 3;
            Ad[rr * 5 + c4] = av[i];
            Bd[rr * 5 + c4] = bv[i];
        }
    }
    __syncthreads();

    const int nIter = 24;
    for (int it = 0; it < nIter; ++it) {
        const int cur = it & 1;
        if (it + 1 < nIter) {
            int k4 = (it + 1) * 4;
#pragma unroll
            for (int i = 0; i < 2; ++i) {
                int e = i * 256 + tid, rr = e >> 2, c4 = e & 3;
                av[i] = Ag[(size_t)(blockM + rr) * 96 + k4 + c4];
                bv[i] = Bg[(size_t)(blockN + rr) * 96 + k4 + c4];
            }
        }
        const unsigned int* Au = (const unsigned int*)lA[cur];
        const unsigned int* Bu = (const unsigned int*)lB[cur];
        Frag a[4], b[2];
#pragma unroll
        for (int fm = 0; fm < 4; ++fm) {
            const unsigned int* p = Au + (waveM * 64 + fm * 16 + l16) * 20 + half * 8;
#pragma unroll
            for (int j = 0; j < 8; ++j) a[fm].u[j] = p[j];
        }
#pragma unroll
        for (int fn = 0; fn < 2; ++fn) {
            const unsigned int* p = Bu + (waveN * 32 + fn * 16 + l16) * 20 + half * 4;
#pragma unroll
            for (int j = 0; j < 8; ++j) b[fn].u[j] = p[(j & 3) + ((j & 4) << 1)];
        }
#pragma unroll
        for (int fm = 0; fm < 4; ++fm)
#pragma unroll
            for (int fn = 0; fn < 2; ++fn)
                acc[fm][fn] = WMMA_BF16(a[fm].v, b[fn].v, acc[fm][fn]);

        if (it + 1 < nIter) {
            u32x4* Ad = (u32x4*)lA[cur ^ 1];
            u32x4* Bd = (u32x4*)lB[cur ^ 1];
#pragma unroll
            for (int i = 0; i < 2; ++i) {
                int e = i * 256 + tid, rr = e >> 2, c4 = e & 3;
                Ad[rr * 5 + c4] = av[i];
                Bd[rr * 5 + c4] = bv[i];
            }
        }
        __syncthreads();
    }

#pragma unroll
    for (int fm = 0; fm < 4; ++fm)
#pragma unroll
        for (int fn = 0; fn < 2; ++fn) {
            int gcol = blockN + waveN * 32 + fn * 16 + l16;
            float bvs = bias[gcol];
#pragma unroll
            for (int r = 0; r < 8; ++r) {
                int grow = blockM + waveM * 64 + fm * 16 + half * 8 + r;
                out[(size_t)grow * 768 + gcol] = acc[fm][fn][r] + bvs;
            }
        }
}

// ===========================================================================
extern "C" void kernel_launch(void* const* d_in, const int* in_sizes, int n_in,
                              void* d_out, int out_size, void* d_ws, size_t ws_size,
                              hipStream_t stream)
{
    (void)in_sizes; (void)n_in; (void)out_size; (void)ws_size;
    const float* x      = (const float*)d_in[0];   // [2,4096,768]
    const float* W_attn = (const float*)d_in[1];   // [768,2304]
    const float* b_attn = (const float*)d_in[2];   // [2304]
    const float* W_proj = (const float*)d_in[3];   // [768,768]
    const float* b_proj = (const float*)d_in[4];   // [768]
    float* out = (float*)d_out;                    // [2,4096,768]

    // workspace (bf16): xbf | Wa^T | Wp^T | qkv | y
    unsigned short* xbf = (unsigned short*)d_ws;                 //  6291456
    unsigned short* Wat = xbf + 6291456ULL;                      //  1769472
    unsigned short* Wpt = Wat + 1769472ULL;                      //   589824
    unsigned short* qkv = Wpt + 589824ULL;                       // 18874368
    unsigned short* yws = qkv + 3ULL * SEL_STRIDE;               //  6291456

    prep_x_kernel<<<dim3(3072), 256, 0, stream>>>(x, xbf);
    prep_w_kernel<<<dim3(864),  256, 0, stream>>>(W_attn, Wat, 768, 2304);
    prep_w_kernel<<<dim3(288),  256, 0, stream>>>(W_proj, Wpt, 768, 768);

    qkv_gemm_kernel<<<dim3(64, 18), 256, 0, stream>>>(xbf, Wat, b_attn, qkv);
    flash_attn_kernel<<<dim3(64, 24), 128, 0, stream>>>(qkv, yws);
    proj_gemm_kernel<<<dim3(64, 6), 256, 0, stream>>>(yws, Wpt, b_proj, out);
}
